// ParallelMLPs_60833916781066
// MI455X (gfx1250) — matrix operand order repaired and checked by reading, not compiled, not run on hardware
//
#include <hip/hip_runtime.h>
#include <hip/hip_bf16.h>

typedef __attribute__((ext_vector_type(16))) __bf16         v16bf;
typedef __attribute__((ext_vector_type(8)))  float          v8f;
typedef __attribute__((ext_vector_type(4)))  float          v4f;
typedef __attribute__((ext_vector_type(8)))  unsigned short v8u16;
typedef __attribute__((ext_vector_type(16))) unsigned short v16u16;

#define B_        256
#define IN_       512
#define O_        10
#define H_        18432
#define M_        128
#define ACT_CHUNK 4608
#define NCOL      32     // hidden columns (neurons) per workgroup
#define KC        32     // K-chunk (bf16 WMMA depth)

// round-to-nearest-even f32 -> bf16 bits
__device__ __forceinline__ unsigned short f2bf(float f) {
    unsigned u = __float_as_uint(f);
    u += 0x7FFFu + ((u >> 16) & 1u);
    return (unsigned short)(u >> 16);
}

// Pre-convert x [256,512] f32 -> bf16 once (avoids per-strip VALU redundancy)
__global__ void cvt_x_bf16(const float* __restrict__ x, unsigned short* __restrict__ xb) {
    int idx = blockIdx.x * blockDim.x + threadIdx.x;
    xb[idx] = f2bf(x[idx]);
}

// out[b,m,o] = out_b[m,o]  (bias broadcast; strips accumulate atomically on top)
__global__ void init_out(const float* __restrict__ out_b, float* __restrict__ out) {
    int idx = blockIdx.x * blockDim.x + threadIdx.x;
    out[idx] = out_b[idx % (M_ * O_)];
}

__global__ __launch_bounds__(256)
void pmlp_main(const unsigned short* __restrict__ xb,   // [256,512] bf16
               const float*          __restrict__ hw,   // [H,512] f32, row-major
               const float*          __restrict__ hb,   // [H]
               const float*          __restrict__ ow,   // [O,H]
               const int*            __restrict__ ids,  // [H]
               float*                __restrict__ out)  // [B,M,O]
{
    __shared__ unsigned short xs[256 * 40];   // x chunk bf16, row stride 40 (pad)
    __shared__ unsigned short ws[32 * 40];    // W chunk bf16, col stride 40 (pad)
    __shared__ float          a_s[256 * 33];  // activated strip, stride 33 (pad)
    __shared__ float          ow_s[O_ * 32];  // out_w strip

    const int t    = threadIdx.x;
    const int lid  = t & 31;
    const int w    = t >> 5;          // wave 0..7
    const int j0   = blockIdx.x * NCOL;

    const int n    = lid & 15;
    const int half = lid >> 4;
    const int ct   = w & 1;           // col-tile (0/1)
    const int rgrp = w >> 1;          // row group: rows rgrp*64 .. +63
    const int col  = ct * 16 + n;     // 0..31 within strip

    v8f c[4];
    #pragma unroll
    for (int i = 0; i < 4; ++i) c[i] = v8f{};

    for (int kc = 0; kc < IN_ / KC; ++kc) {
        const int k0 = kc * KC;
        __syncthreads();
        // ---- stage x chunk: 256 rows x 32 K (bf16 copy), 16B per lane per pass
        #pragma unroll
        for (int p = 0; p < 4; ++p) {
            int row  = p * 64 + (t >> 2);
            int koff = (t & 3) * 8;
            *(v8u16*)&xs[row * 40 + koff] =
                *(const v8u16*)&xb[row * IN_ + k0 + koff];
        }
        // ---- stage W chunk: 32 cols x 32 K, f32 -> bf16
        {
            int wc = t >> 3;           // 0..31
            int wk = (t & 7) * 4;      // 0..28
            const float* wp = &hw[(size_t)(j0 + wc) * IN_ + k0 + wk];
            v4f v = *(const v4f*)wp;
            if (kc + 1 < IN_ / KC)                 // prefetch next K-chunk of W
                __builtin_prefetch(wp + KC, 0, 0); // -> global_prefetch_b8
            uint2 pk;
            pk.x = (unsigned)f2bf(v.x) | ((unsigned)f2bf(v.y) << 16);
            pk.y = (unsigned)f2bf(v.z) | ((unsigned)f2bf(v.w) << 16);
            *(uint2*)&ws[wc * 40 + wk] = pk;
        }
        __syncthreads();

        // ---- B fragment (32x16): lane holds col=ct*16+n, K = 8*half+0..7 | 16+8*half+0..7
        v8u16 blo = *(const v8u16*)&ws[col * 40 + half * 8];
        v8u16 bhi = *(const v8u16*)&ws[col * 40 + 16 + half * 8];
        v16u16 bu = __builtin_shufflevector(blo, bhi,
                      0,1,2,3,4,5,6,7,8,9,10,11,12,13,14,15);
        v16bf bf = __builtin_bit_cast(v16bf, bu);

        // ---- 4 row tiles: A frag lane holds row r0+n, K = 8*half+0..7 | 16+8*half+0..7
        #pragma unroll
        for (int i = 0; i < 4; ++i) {
            int r0 = rgrp * 64 + i * 16;
            const unsigned short* ap = &xs[(r0 + n) * 40];
            v8u16 alo = *(const v8u16*)&ap[8 * half];
            v8u16 ahi = *(const v8u16*)&ap[16 + 8 * half];
            v16u16 au = __builtin_shufflevector(alo, ahi,
                          0,1,2,3,4,5,6,7,8,9,10,11,12,13,14,15);
            v16bf af = __builtin_bit_cast(v16bf, au);
            c[i] = __builtin_amdgcn_wmma_f32_16x16x32_bf16(
                       false, af, false, bf, (short)0, c[i], false, false);
        }
    }

    // ---- bias + activation (uniform per strip), scatter to a_s
    // cheap branchless forms: sigmoid = rcp(1+exp(-v)); tanh = 2*sigmoid(2v)-1
    {
        const float hbv   = hb[j0 + col];
        const int  actsel = j0 / ACT_CHUNK;  // 0:relu 1:tanh 2:sigmoid 3:identity
        const bool isTanh = (actsel == 1);
        #pragma unroll
        for (int i = 0; i < 4; ++i) {
            int rbase = rgrp * 64 + i * 16 + 8 * half;  // C: VGPR r -> row rbase+r
            #pragma unroll
            for (int r = 0; r < 8; ++r) {
                float v  = c[i][r] + hbv;
                float av;
                if (actsel == 0) {
                    av = fmaxf(v, 0.f);
                } else if (actsel == 3) {
                    av = v;
                } else {
                    float z = isTanh ? (-2.f * v) : (-v);
                    float s = __builtin_amdgcn_rcpf(1.f + __expf(z));
                    av = isTanh ? __builtin_fmaf(2.f, s, -1.f) : s;
                }
                a_s[(rbase + r) * 33 + col] = av;
            }
        }
    }
    // ---- stage out_w strip [O x 32]
    for (int idx = t; idx < O_ * 32; idx += 256) {
        int o = idx >> 5, j = idx & 31;
        ow_s[o * 32 + j] = ow[(size_t)o * H_ + j0 + j];
    }
    __syncthreads();

    // ---- strip lies in exactly one model segment (all boundaries %32==0)
    const int m = ids[j0];
    float acc[O_];
    #pragma unroll
    for (int o = 0; o < O_; ++o) acc[o] = 0.f;
    #pragma unroll 4
    for (int j = 0; j < 32; ++j) {
        float av = a_s[t * 33 + j];
        #pragma unroll
        for (int o = 0; o < O_; ++o) acc[o] += av * ow_s[o * 32 + j];
    }
    float* op = out + (size_t)t * (M_ * O_) + m * O_;
    #pragma unroll
    for (int o = 0; o < O_; ++o) atomicAdd(&op[o], acc[o]);
}

extern "C" void kernel_launch(void* const* d_in, const int* in_sizes, int n_in,
                              void* d_out, int out_size, void* d_ws, size_t ws_size,
                              hipStream_t stream) {
    (void)in_sizes; (void)n_in; (void)out_size; (void)ws_size;
    const float* x   = (const float*)d_in[0];
    const float* hw  = (const float*)d_in[1];
    const float* hb  = (const float*)d_in[2];
    const float* ow  = (const float*)d_in[3];
    const float* ob  = (const float*)d_in[4];
    const int*   ids = (const int*)d_in[5];
    float* out = (float*)d_out;
    unsigned short* xb = (unsigned short*)d_ws;   // 256KB bf16 x

    cvt_x_bf16<<<(B_ * IN_) / 256, 256, 0, stream>>>(x, xb);
    init_out<<<(B_ * M_ * O_) / 256, 256, 0, stream>>>(ob, out);
    pmlp_main<<<H_ / NCOL, 256, 0, stream>>>(xb, hw, hb, ow, ids, out);
}
